// TemporalSpatialFusion_12068858101762
// MI455X (gfx1250) — matrix operand order repaired and checked by reading, not compiled, not run on hardware
//
#include <hip/hip_runtime.h>

// ---------------------------------------------------------------------------
// Problem constants (match reference)
// ---------------------------------------------------------------------------
#define BB   4
#define SS   1024
#define DD   256
#define HH   8
#define DK   32
#define EE   8192
#define FDIM 256
#define FH   8
#define SCALE_INV_SQDK 0.17677669529663687f  // 1/sqrt(32)

// GEMM tiling
#define BM 128
#define BN 64
#define BK 64
// LDS row pad (ushorts): 88*2 = 176B = 16B-aligned rows; bank stride 44 -> the 16
// fragment lanes start on 16 distinct bank groups -> conflict-free ds_load_b128.
#define APAD (BK + 24)
#define PPAD 40         // flash P-tile padded row: 80B, aligned + conflict-free

// ---------------------------------------------------------------------------
// CDNA5 WMMA types / helpers
// ---------------------------------------------------------------------------
typedef __attribute__((ext_vector_type(16))) __bf16 v16bf;
typedef __attribute__((ext_vector_type(8)))  float  v8f;

union FragBF {
    unsigned short u[16];
    unsigned int   w[8];
    v16bf          v;
};

__device__ inline v8f wmma_bf16(v16bf a, v16bf b, v8f c) {
    return __builtin_amdgcn_wmma_f32_16x16x32_bf16(false, a, false, b, (short)0, c,
                                                   false, false);
}

// float -> bf16 bits via real fptrunc (backend picks v_cvt_pk_bf16_f32)
__device__ inline unsigned short bfbits(float f) {
    union { __bf16 h; unsigned short u; } cv;
    cv.h = (__bf16)f;
    return cv.u;
}

// CDNA5 async global->LDS copy (ASYNCcnt-tracked); generic LDS pointer's low
// 32 bits are the wave-relative LDS byte address.
__device__ inline void async_g2l_b128(void* lds, const void* gsrc) {
    unsigned loff = (unsigned)(unsigned long long)lds;
    asm volatile("global_load_async_to_lds_b128 %0, %1, off"
                 :: "v"(loff), "v"(gsrc)
                 : "memory");
}
__device__ inline void wait_async0() {
    asm volatile("s_wait_asynccnt 0" ::: "memory");
}

__device__ inline float gelu_tanh(float x) {
    float x3 = x * x * x;
    return 0.5f * x * (1.0f + tanhf(0.7978845608028654f * (x + 0.044715f * x3)));
}

// order-preserving float<->uint encode for atomicMax
__device__ inline unsigned fenc(float f) {
    unsigned u = __float_as_uint(f);
    return (u & 0x80000000u) ? ~u : (u | 0x80000000u);
}
__device__ inline float fdec(unsigned e) {
    unsigned u = (e & 0x80000000u) ? (e & 0x7fffffffu) : ~e;
    return __uint_as_float(u);
}

// ---------------------------------------------------------------------------
// Weight pre-conversion: fp32 -> bf16 (RNE), vectorized
// ---------------------------------------------------------------------------
__global__ __launch_bounds__(256)
void cvt_f32_bf16_k(const float* __restrict__ X, unsigned short* __restrict__ Y,
                    size_t n4)
{
    size_t t = (size_t)blockIdx.x * 256 + threadIdx.x;
    if (t >= n4) return;
    float4 v = ((const float4*)X)[t];
    unsigned lo = (unsigned)bfbits(v.x) | ((unsigned)bfbits(v.y) << 16);
    unsigned hi = (unsigned)bfbits(v.z) | ((unsigned)bfbits(v.w) << 16);
    ((uint2*)Y)[t] = make_uint2(lo, hi);
}

// ---------------------------------------------------------------------------
// Generic WMMA GEMM:  Y[M,N] = act(X[M,K] @ W[N,K]^T + bias[N]) (+ resid[M,N])
// X is fp32 (converted while staging); W is pre-converted bf16 and staged into
// LDS via global_load_async_to_lds_b128.
// Block: 256 threads (8 waves), tile 128x64, K-step 64 -> 8 WMMAs/wave/K-step.
// ---------------------------------------------------------------------------
__global__ __launch_bounds__(256)
void gemm_bias_act(const float* __restrict__ X, const unsigned short* __restrict__ Wbf,
                   const float* __restrict__ bias, const float* __restrict__ resid,
                   float* __restrict__ Y, int M, int N, int K, int act)
{
    __shared__ __align__(16) unsigned short As[BM][APAD];
    __shared__ __align__(16) unsigned short Bs[BN][APAD];

    const int tid  = threadIdx.x;
    const int lane = tid & 31;
    const int wave = tid >> 5;
    const int hl   = lane >> 4;
    const int l15  = lane & 15;

    const int m0 = blockIdx.y * BM;
    const int n0 = blockIdx.x * BN;

    const int mt = wave >> 1;  // 0..3 -> 32-row group
    const int np = wave & 1;   // 0..1 -> 32-col group

    v8f acc[2][2];
#pragma unroll
    for (int i = 0; i < 2; ++i)
#pragma unroll
        for (int j = 0; j < 2; ++j)
            acc[i][j] = (v8f){0, 0, 0, 0, 0, 0, 0, 0};

    for (int k0 = 0; k0 < K; k0 += BK) {
        // B tile: async copy 64 rows x 64 bf16 straight into LDS (no conversion)
#pragma unroll
        for (int i = 0; i < 2; ++i) {
            int f  = tid + i * 256;  // 0..511
            int r  = f >> 3;         // row 0..63
            int c8 = f & 7;          // 16B chunk
            int gn = n0 + r;
            if (gn < N) {
                async_g2l_b128(&Bs[r][c8 * 8],
                               Wbf + (size_t)gn * K + k0 + c8 * 8);
            } else {
                ((uint2*)&Bs[r][0])[c8 * 2 + 0] = make_uint2(0u, 0u);
                ((uint2*)&Bs[r][0])[c8 * 2 + 1] = make_uint2(0u, 0u);
            }
        }

        // prefetch next A tile into cache (global_prefetch_b8)
        if (k0 + BK < K) {
            int pr = m0 + (tid >> 1);
            if (pr >= M) pr = M - 1;
            __builtin_prefetch(X + (size_t)pr * K + k0 + BK + (tid & 1) * 32, 0, 1);
        }

        // A tile: fp32 load -> v_cvt_pk_bf16_f32 -> ds_store_b64
#pragma unroll
        for (int i = 0; i < 8; ++i) {
            int f  = tid + i * 256;  // 0..2047
            int r  = f >> 4;
            int c4 = f & 15;
            int gm = m0 + r;
            float4 va = make_float4(0.f, 0.f, 0.f, 0.f);
            if (gm < M) va = ((const float4*)(X + (size_t)gm * K + k0))[c4];
            unsigned lo = (unsigned)bfbits(va.x) | ((unsigned)bfbits(va.y) << 16);
            unsigned hi = (unsigned)bfbits(va.z) | ((unsigned)bfbits(va.w) << 16);
            ((uint2*)&As[r][0])[c4] = make_uint2(lo, hi);
        }

        wait_async0();
        __syncthreads();

#pragma unroll
        for (int kk = 0; kk < BK; kk += 32) {
            FragBF a[2], b[2];
#pragma unroll
            for (int i = 0; i < 2; ++i) {
                int arow = mt * 32 + i * 16 + l15;
                uint4 t0 = *(const uint4*)&As[arow][kk + hl * 8];
                uint4 t1 = *(const uint4*)&As[arow][kk + 16 + hl * 8];
                a[i].w[0] = t0.x; a[i].w[1] = t0.y; a[i].w[2] = t0.z; a[i].w[3] = t0.w;
                a[i].w[4] = t1.x; a[i].w[5] = t1.y; a[i].w[6] = t1.z; a[i].w[7] = t1.w;
            }
#pragma unroll
            for (int j = 0; j < 2; ++j) {
                int brow = np * 32 + j * 16 + l15;
                uint4 t0 = *(const uint4*)&Bs[brow][kk + hl * 8];
                uint4 t1 = *(const uint4*)&Bs[brow][kk + 16 + hl * 8];
                b[j].w[0] = t0.x; b[j].w[1] = t0.y; b[j].w[2] = t0.z; b[j].w[3] = t0.w;
                b[j].w[4] = t1.x; b[j].w[5] = t1.y; b[j].w[6] = t1.z; b[j].w[7] = t1.w;
            }
#pragma unroll
            for (int i = 0; i < 2; ++i)
#pragma unroll
                for (int j = 0; j < 2; ++j)
                    acc[i][j] = wmma_bf16(a[i].v, b[j].v, acc[i][j]);
        }
        __syncthreads();
    }

    // epilogue
#pragma unroll
    for (int i = 0; i < 2; ++i)
#pragma unroll
        for (int j = 0; j < 2; ++j)
#pragma unroll
            for (int e = 0; e < 8; ++e) {
                int gm = m0 + mt * 32 + i * 16 + e + 8 * hl;
                int gn = n0 + np * 32 + j * 16 + l15;
                if (gm < M && gn < N) {
                    float r0 = acc[i][j][e];
                    if (bias) r0 += bias[gn];
                    if (act == 1) r0 = gelu_tanh(r0);
                    if (resid) r0 += resid[(size_t)gm * N + gn];
                    Y[(size_t)gm * N + gn] = r0;
                }
            }
}

// ---------------------------------------------------------------------------
// LayerNorm: one block per row
// ---------------------------------------------------------------------------
__global__ __launch_bounds__(256)
void layernorm_k(const float* __restrict__ X, const float* __restrict__ g,
                 const float* __restrict__ b, float* __restrict__ Y, int D)
{
    __shared__ float red[256];
    const int row = blockIdx.x;
    const float* xr = X + (size_t)row * D;

    float s = 0.f;
    for (int c = threadIdx.x; c < D; c += 256) s += xr[c];
    red[threadIdx.x] = s;
    __syncthreads();
    for (int o = 128; o > 0; o >>= 1) {
        if (threadIdx.x < o) red[threadIdx.x] += red[threadIdx.x + o];
        __syncthreads();
    }
    float mean = red[0] / (float)D;
    __syncthreads();

    float v = 0.f;
    for (int c = threadIdx.x; c < D; c += 256) {
        float d = xr[c] - mean;
        v += d * d;
    }
    red[threadIdx.x] = v;
    __syncthreads();
    for (int o = 128; o > 0; o >>= 1) {
        if (threadIdx.x < o) red[threadIdx.x] += red[threadIdx.x + o];
        __syncthreads();
    }
    float rstd = rsqrtf(red[0] / (float)D + 1e-5f);

    for (int c = threadIdx.x; c < D; c += 256)
        Y[(size_t)row * D + c] = (xr[c] - mean) * rstd * g[c] + b[c];
}

// ---------------------------------------------------------------------------
// Flash attention, 1 wave per block, 16-query tile, 32-key chunks, DK=32.
// Q:(B,qRowsPerB,256) head-major cols, K/V:(B,kRowsPerB,256), O:(B,oRowsPerB,256)
// maskMode 0: dense.  maskMode 1: longformer band(+-256) + global(first 16).
// ---------------------------------------------------------------------------
__global__ __launch_bounds__(32)
void flash_attn_k(const float* __restrict__ Q, const float* __restrict__ Kb,
                  const float* __restrict__ Vb, float* __restrict__ O,
                  int qRowsPerB, int oRowsPerB, int kRowsPerB,
                  int Sk, float scale, int maskMode)
{
    const int lane = threadIdx.x;
    const int hl   = lane >> 4;
    const int l15  = lane & 15;
    const int i0   = blockIdx.x * 16;
    const int h    = blockIdx.y;
    const int b    = blockIdx.z;
    const int D    = 256;

    __shared__ __align__(16) unsigned short Pbuf[16][PPAD];

    // Q A-fragment (rows i0..i0+15, k = head dim 0..31)
    FragBF qa;
    {
        const float* qrow = Q + ((size_t)b * qRowsPerB + i0 + l15) * D + h * DK;
#pragma unroll
        for (int i = 0; i < 16; ++i) {
            int k = ((i >> 3) << 4) + hl * 8 + (i & 7);
            qa.u[i] = bfbits(qrow[k]);
        }
    }

    v8f o0 = {0, 0, 0, 0, 0, 0, 0, 0};
    v8f o1 = {0, 0, 0, 0, 0, 0, 0, 0};
    float m[8], l[8];
#pragma unroll
    for (int e = 0; e < 8; ++e) { m[e] = -1e30f; l[e] = 0.f; }

    for (int j0 = 0; j0 < Sk; j0 += 32) {
        if (maskMode == 1) {
            bool need = (j0 == 0) || (i0 == 0) ||
                        ((j0 + 31 >= i0 - 256) && (j0 <= i0 + 271));
            if (!need) continue;
        }

        // K^T B-fragments for 32 keys
        FragBF kb0, kb1;
#pragma unroll
        for (int i = 0; i < 16; ++i) {
            int d = ((i >> 3) << 4) + hl * 8 + (i & 7);
            kb0.u[i] = bfbits(Kb[((size_t)b * kRowsPerB + j0 + l15) * D + h * DK + d]);
            kb1.u[i] = bfbits(Kb[((size_t)b * kRowsPerB + j0 + 16 + l15) * D + h * DK + d]);
        }
        v8f z = {0, 0, 0, 0, 0, 0, 0, 0};
        v8f s0 = wmma_bf16(qa.v, kb0.v, z);
        v8f s1 = wmma_bf16(qa.v, kb1.v, z);

        float sv0[8], sv1[8];
#pragma unroll
        for (int e = 0; e < 8; ++e) {
            int   qi = i0 + e + 8 * hl;
            int   ja = j0 + l15;
            int   jb = j0 + 16 + l15;
            float a0 = s0[e] * scale;
            float a1 = s1[e] * scale;
            if (maskMode == 1) {
                int d0 = qi - ja; if (d0 < 0) d0 = -d0;
                int d1 = qi - jb; if (d1 < 0) d1 = -d1;
                bool ok0 = (qi < 16) || (ja < 16) || (d0 <= 256);
                bool ok1 = (qi < 16) || (jb < 16) || (d1 <= 256);
                if (!ok0) a0 = -1e9f;
                if (!ok1) a1 = -1e9f;
            }
            sv0[e] = a0;
            sv1[e] = a1;
        }

        // online softmax update
#pragma unroll
        for (int e = 0; e < 8; ++e) {
            float cm = fmaxf(sv0[e], sv1[e]);
            for (int off = 1; off < 16; off <<= 1)
                cm = fmaxf(cm, __shfl_xor(cm, off, 32));
            float mn = fmaxf(m[e], cm);
            float c  = __expf(m[e] - mn);
            float p0 = (sv0[e] <= -1e8f) ? 0.f : __expf(sv0[e] - mn);
            float p1 = (sv1[e] <= -1e8f) ? 0.f : __expf(sv1[e] - mn);
            float ps = p0 + p1;
            for (int off = 1; off < 16; off <<= 1)
                ps += __shfl_xor(ps, off, 32);
            l[e] = l[e] * c + ps;
            m[e] = mn;
            o0[e] *= c;
            o1[e] *= c;
            Pbuf[e + 8 * hl][l15]      = bfbits(p0);
            Pbuf[e + 8 * hl][16 + l15] = bfbits(p1);
        }
        __syncthreads();

        // P A-fragment
        FragBF pa;
#pragma unroll
        for (int i = 0; i < 16; ++i) {
            int k = ((i >> 3) << 4) + hl * 8 + (i & 7);
            pa.u[i] = Pbuf[l15][k];
        }
        // V B-fragments (32 keys x 32 dims -> two 16-col tiles)
        FragBF vb0, vb1;
#pragma unroll
        for (int i = 0; i < 16; ++i) {
            size_t vr = ((size_t)b * kRowsPerB + j0 + ((i >> 3) << 4) + hl * 8 + (i & 7)) * D + h * DK;
            vb0.u[i] = bfbits(Vb[vr + l15]);
            vb1.u[i] = bfbits(Vb[vr + 16 + l15]);
        }
        o0 = wmma_bf16(pa.v, vb0.v, o0);
        o1 = wmma_bf16(pa.v, vb1.v, o1);
        __syncthreads();
    }

#pragma unroll
    for (int e = 0; e < 8; ++e) {
        int    qi  = i0 + e + 8 * hl;
        float  inv = 1.0f / l[e];
        size_t orw = ((size_t)b * oRowsPerB + qi) * 256 + h * DK;
        O[orw + l15]      = o0[e] * inv;
        O[orw + 16 + l15] = o1[e] * inv;
    }
}

// ---------------------------------------------------------------------------
// HGT edge pass A: sc = (k_e @ W_att[h] . q_e) * mu[h] / sqrt(DK); atomicMax
// ---------------------------------------------------------------------------
__global__ __launch_bounds__(256)
void hgt_score_k(const float* __restrict__ Ksrc, const float* __restrict__ Qdst,
                 const int* __restrict__ eidx, const float* __restrict__ watt,
                 const float* __restrict__ mu, float* __restrict__ sc,
                 unsigned* __restrict__ mx, int nSrc, int nDst)
{
    int t = blockIdx.x * 256 + threadIdx.x;
    if (t >= BB * EE * HH) return;
    int h = t & 7;
    int e = (t >> 3) & (EE - 1);
    int b = t / (EE * HH);
    int si = eidx[e];
    int di = eidx[EE + e];

    const float* ke = Ksrc + ((size_t)(b * nSrc + si)) * DD + h * DK;
    const float* qe = Qdst + ((size_t)(b * nDst + di)) * DD + h * DK;
    const float* W  = watt + h * DK * DK;

    float tmp[DK];
#pragma unroll
    for (int j = 0; j < DK; ++j) tmp[j] = 0.f;
    for (int i = 0; i < DK; ++i) {
        float kv = ke[i];
#pragma unroll
        for (int j = 0; j < DK; ++j) tmp[j] += kv * W[i * DK + j];
    }
    float sv = 0.f;
#pragma unroll
    for (int j = 0; j < DK; ++j) sv += tmp[j] * qe[j];
    sv = sv * mu[h] * SCALE_INV_SQDK;

    sc[t] = sv;
    atomicMax(&mx[(size_t)(b * nDst + di) * HH + h], fenc(sv));
}

// ---------------------------------------------------------------------------
// HGT edge pass B: ex = exp(sc - max); den += ex; agg += ex * (v_e @ W_msg[h])
// ---------------------------------------------------------------------------
__global__ __launch_bounds__(256)
void hgt_accum_k(const float* __restrict__ Vsrc, const int* __restrict__ eidx,
                 const float* __restrict__ wmsg, const float* __restrict__ sc,
                 const unsigned* __restrict__ mx, float* __restrict__ den,
                 float* __restrict__ agg, int nSrc, int nDst)
{
    int t = blockIdx.x * 256 + threadIdx.x;
    if (t >= BB * EE * HH) return;
    int h = t & 7;
    int e = (t >> 3) & (EE - 1);
    int b = t / (EE * HH);
    int si = eidx[e];
    int di = eidx[EE + e];

    float mxv = fdec(mx[(size_t)(b * nDst + di) * HH + h]);
    float ex  = __expf(sc[t] - mxv);
    atomicAdd(&den[(size_t)(b * nDst + di) * HH + h], ex);

    const float* ve = Vsrc + ((size_t)(b * nSrc + si)) * DD + h * DK;
    const float* W  = wmsg + h * DK * DK;
    float msg[DK];
#pragma unroll
    for (int j = 0; j < DK; ++j) msg[j] = 0.f;
    for (int i = 0; i < DK; ++i) {
        float vv = ve[i];
#pragma unroll
        for (int j = 0; j < DK; ++j) msg[j] += vv * W[i * DK + j];
    }
    float* ag = agg + ((size_t)(b * nDst + di)) * DD + h * DK;
#pragma unroll
    for (int j = 0; j < DK; ++j) atomicAdd(&ag[j], ex * msg[j]);
}

// ---------------------------------------------------------------------------
// HGT finalize: gbuf = gelu(agg / (den + 1e-9))
// ---------------------------------------------------------------------------
__global__ __launch_bounds__(256)
void hgt_fin_k(const float* __restrict__ agg, const float* __restrict__ den,
               float* __restrict__ gbuf, size_t total)
{
    size_t t = (size_t)blockIdx.x * 256 + threadIdx.x;
    if (t >= total) return;
    int col  = (int)(t & (DD - 1));
    int h    = col >> 5;
    size_t n = t >> 8;
    float d  = den[n * HH + h] + 1e-9f;
    gbuf[t]  = gelu_tanh(agg[t] / d);
}

// ---------------------------------------------------------------------------
// misc elementwise
// ---------------------------------------------------------------------------
__global__ __launch_bounds__(256)
void fill_zero_k(unsigned* __restrict__ p, size_t n)
{
    size_t t = (size_t)blockIdx.x * 256 + threadIdx.x;
    if (t < n) p[t] = 0u;
}

__global__ __launch_bounds__(256)
void concat2_k(const float* __restrict__ A, const float* __restrict__ Bv,
               float* __restrict__ Y, size_t rows)
{
    size_t t = (size_t)blockIdx.x * 256 + threadIdx.x;
    if (t >= rows * 512) return;
    size_t r = t >> 9;
    int    c = (int)(t & 511);
    Y[t] = (c < 256) ? A[r * 256 + c] : Bv[r * 256 + (c - 256)];
}

// ---------------------------------------------------------------------------
// host-side launcher
// ---------------------------------------------------------------------------
extern "C" void kernel_launch(void* const* d_in, const int* in_sizes, int n_in,
                              void* d_out, int out_size, void* d_ws, size_t ws_size,
                              hipStream_t stream)
{
    (void)in_sizes; (void)n_in; (void)out_size; (void)ws_size;

    auto F = [&](int i) { return (const float*)d_in[i]; };
    auto I = [&](int i) { return (const int*)d_in[i]; };

    const int NN[4] = {2048, 1024, 512, 64};  // alert, ip, host, tactic

    // ---- workspace layout (floats) ----
    float* WSP = (float*)d_ws;
    size_t off = 0;
    auto alloc = [&](size_t n) { float* p = WSP + off; off += n; return p; };

    float* xb = alloc((size_t)BB * SS * DD);        // temporal stream (persistent)
    float* fe[4];
    for (int t = 0; t < 4; ++t) fe[t] = alloc((size_t)BB * NN[t] * DD);  // persistent
    float* wcvt = alloc(131072);  // bf16 weight staging (max 1024x256 elems)
    unsigned short* wbf = (unsigned short*)wcvt;
    const size_t scratch0 = off;

    auto gemm = [&](const float* X, const float* Wf, const float* bias,
                    const float* resid, float* Y, int M, int N, int K, int act) {
        size_t n4 = (size_t)N * K / 4;
        cvt_f32_bf16_k<<<dim3((unsigned)((n4 + 255) / 256)), dim3(256), 0, stream>>>(
            Wf, wbf, n4);
        dim3 g((N + BN - 1) / BN, (M + BM - 1) / BM);
        gemm_bias_act<<<g, dim3(256), 0, stream>>>(X, wbf, bias, resid, Y, M, N, K, act);
    };
    auto lnorm = [&](const float* X, const float* g, const float* b, float* Y,
                     int rows, int D) {
        layernorm_k<<<dim3(rows), dim3(256), 0, stream>>>(X, g, b, Y, D);
    };
    auto fill0 = [&](void* p, size_t n) {
        fill_zero_k<<<dim3((unsigned)((n + 255) / 256)), dim3(256), 0, stream>>>(
            (unsigned*)p, n);
    };

    // =====================================================================
    // Phase 1: temporal transformer stack
    // =====================================================================
    off = scratch0;
    float* lnb  = alloc((size_t)BB * SS * DD);
    float* qb   = alloc((size_t)BB * SS * DD);
    float* kb   = alloc((size_t)BB * SS * DD);
    float* vb   = alloc((size_t)BB * SS * DD);
    float* attn = alloc((size_t)BB * SS * DD);
    float* ffh  = alloc((size_t)BB * SS * 4 * DD);

    hipMemcpyAsync(xb, d_in[0], (size_t)BB * SS * DD * sizeof(float),
                   hipMemcpyDeviceToDevice, stream);

    const int Mt = BB * SS;  // 4096
    for (int L = 0; L < 4; ++L) {
        int base = 12 + 16 * L;
        lnorm(xb, F(base + 0), F(base + 1), lnb, Mt, DD);
        gemm(lnb, F(base + 4), F(base + 5), nullptr, qb, Mt, DD, DD, 0);
        gemm(lnb, F(base + 6), F(base + 7), nullptr, kb, Mt, DD, DD, 0);
        gemm(lnb, F(base + 8), F(base + 9), nullptr, vb, Mt, DD, DD, 0);
        flash_attn_k<<<dim3(SS / 16, HH, BB), dim3(32), 0, stream>>>(
            qb, kb, vb, attn, SS, SS, SS, SS, SCALE_INV_SQDK, 1);
        gemm(attn, F(base + 10), F(base + 11), xb, xb, Mt, DD, DD, 0);
        lnorm(xb, F(base + 2), F(base + 3), lnb, Mt, DD);
        gemm(lnb, F(base + 12), F(base + 13), nullptr, ffh, Mt, 4 * DD, DD, 1);
        gemm(ffh, F(base + 14), F(base + 15), xb, xb, Mt, DD, 4 * DD, 0);
    }

    // =====================================================================
    // Phase 2: HGT spatial stack
    // =====================================================================
    off = scratch0;
    size_t totN = 0;
    size_t koff[4];
    for (int t = 0; t < 4; ++t) { koff[t] = totN * DD; totN += (size_t)BB * NN[t]; }
    float* kall = alloc(totN * DD);
    float* qall = alloc(totN * DD);
    float* vall = alloc(totN * DD);
    float* mxb  = alloc((size_t)BB * 2048 * HH);   // reused per type (max size)
    float* denb = alloc((size_t)BB * 2048 * HH);
    float* aggb = alloc((size_t)BB * 2048 * DD);
    float* gbuf = alloc((size_t)BB * 2048 * DD);
    float* scb  = alloc(3 * (size_t)BB * EE * HH);

    for (int t = 0; t < 4; ++t)
        hipMemcpyAsync(fe[t], d_in[2 + t], (size_t)BB * NN[t] * DD * sizeof(float),
                       hipMemcpyDeviceToDevice, stream);

    // relations: gr -> (srcType, dstType); types: 0 alert 1 ip 2 host 3 tactic
    const int relSrc[6] = {0, 1, 0, 2, 0, 3};
    const int relDst[6] = {1, 0, 2, 0, 3, 0};

    const int edgeBlocks = (BB * EE * HH + 255) / 256;
    for (int L = 0; L < 2; ++L) {
        int sb = 76 + 50 * L;
        for (int t = 0; t < 4; ++t) {
            int rows = BB * NN[t];
            gemm(fe[t], F(sb + 2 * t),      F(sb + 2 * t + 1),      nullptr,
                 kall + koff[t], rows, DD, DD, 0);
            gemm(fe[t], F(sb + 8 + 2 * t),  F(sb + 8 + 2 * t + 1),  nullptr,
                 qall + koff[t], rows, DD, DD, 0);
            gemm(fe[t], F(sb + 16 + 2 * t), F(sb + 16 + 2 * t + 1), nullptr,
                 vall + koff[t], rows, DD, DD, 0);
        }
        for (int t = 0; t < 4; ++t) {
            size_t nNodes = (size_t)BB * NN[t];
            fill0(mxb, nNodes * HH);
            fill0(denb, nNodes * HH);
            fill0(aggb, nNodes * DD);
            int slot = 0;
            for (int gr = 0; gr < 6; ++gr) {
                if (relDst[gr] != t) continue;
                int s = relSrc[gr];
                hgt_score_k<<<dim3(edgeBlocks), dim3(256), 0, stream>>>(
                    kall + koff[s], qall + koff[t], I(6 + gr),
                    F(sb + 32 + gr), F(sb + 44 + gr),
                    scb + (size_t)slot * BB * EE * HH, (unsigned*)mxb,
                    NN[s], NN[t]);
                ++slot;
            }
            slot = 0;
            for (int gr = 0; gr < 6; ++gr) {
                if (relDst[gr] != t) continue;
                int s = relSrc[gr];
                hgt_accum_k<<<dim3(edgeBlocks), dim3(256), 0, stream>>>(
                    vall + koff[s], I(6 + gr), F(sb + 38 + gr),
                    scb + (size_t)slot * BB * EE * HH, (const unsigned*)mxb,
                    denb, aggb, NN[s], NN[t]);
                ++slot;
            }
            size_t total = nNodes * DD;
            hgt_fin_k<<<dim3((unsigned)((total + 255) / 256)), dim3(256), 0, stream>>>(
                aggb, denb, gbuf, total);
            gemm(gbuf, F(sb + 24 + 2 * t), F(sb + 24 + 2 * t + 1), fe[t], fe[t],
                 BB * NN[t], DD, DD, 0);
        }
    }

    // =====================================================================
    // Phase 3: fusion
    // =====================================================================
    off = scratch0;
    float* tf   = alloc((size_t)BB * SS * FDIM);
    float* sf   = alloc((size_t)BB * 2048 * FDIM);
    float* qts  = alloc((size_t)BB * SS * FDIM);
    float* kts  = alloc((size_t)BB * 2048 * FDIM);
    float* vts  = alloc((size_t)BB * 2048 * FDIM);
    float* ots  = alloc((size_t)BB * SS * FDIM);
    float* tatt = alloc((size_t)BB * SS * FDIM);
    float* qst  = alloc((size_t)BB * 2048 * FDIM);
    float* kst  = alloc((size_t)BB * SS * FDIM);
    float* vst  = alloc((size_t)BB * SS * FDIM);
    float* ost  = alloc((size_t)BB * SS * FDIM);
    float* satt = alloc((size_t)BB * SS * FDIM);
    float* comb = alloc((size_t)BB * SS * 2 * FDIM);
    float* lnc  = alloc((size_t)BB * SS * 2 * FDIM);

    const int Ma = BB * 2048;  // 8192 alert rows
    gemm(xb,    F(176), F(177), nullptr, tf, Mt, FDIM, DD, 0);
    gemm(fe[0], F(178), F(179), nullptr, sf, Ma, FDIM, DD, 0);

    // mha_ts: q from tf, k/v from sf (packed in_proj rows [0,256),[256,512),[512,768))
    gemm(tf, F(180),                F(181),       nullptr, qts, Mt, FDIM, FDIM, 0);
    gemm(sf, F(180) + 256 * FDIM,   F(181) + 256, nullptr, kts, Ma, FDIM, FDIM, 0);
    gemm(sf, F(180) + 512 * FDIM,   F(181) + 512, nullptr, vts, Ma, FDIM, FDIM, 0);
    flash_attn_k<<<dim3(SS / 16, FH, BB), dim3(32), 0, stream>>>(
        qts, kts, vts, ots, SS, SS, 2048, 2048, SCALE_INV_SQDK, 0);
    gemm(ots, F(182), F(183), nullptr, tatt, Mt, FDIM, FDIM, 0);

    // mha_st: q from sf, k/v from tf; only first S query rows are consumed
    gemm(sf, F(184),              F(185),       nullptr, qst, Ma, FDIM, FDIM, 0);
    gemm(tf, F(184) + 256 * FDIM, F(185) + 256, nullptr, kst, Mt, FDIM, FDIM, 0);
    gemm(tf, F(184) + 512 * FDIM, F(185) + 512, nullptr, vst, Mt, FDIM, FDIM, 0);
    flash_attn_k<<<dim3(SS / 16, FH, BB), dim3(32), 0, stream>>>(
        qst, kst, vst, ost, 2048, SS, SS, SS, SCALE_INV_SQDK, 0);
    gemm(ost, F(186), F(187), nullptr, satt, Mt, FDIM, FDIM, 0);

    size_t ctot = (size_t)Mt * 2 * FDIM;
    concat2_k<<<dim3((unsigned)((ctot + 255) / 256)), dim3(256), 0, stream>>>(
        tatt, satt, comb, (size_t)Mt);
    lnorm(comb, F(188), F(189), lnc, Mt, 2 * FDIM);
    gemm(lnc, F(190), F(191), nullptr, (float*)d_out, Mt, FDIM, 2 * FDIM, 0);
}
